// CRQVAE_76510547410981
// MI455X (gfx1250) — matrix operand order repaired and checked by reading, not compiled, not run on hardware
//
#include <hip/hip_runtime.h>
#include <hip/hip_bf16.h>

// ---------------------------------------------------------------------------
// Types for CDNA5 WMMA (wave32): v_wmma_f32_16x16x32_bf16
// ---------------------------------------------------------------------------
typedef __bf16 bf16;
typedef __attribute__((ext_vector_type(16))) __bf16 v16bf;
typedef __attribute__((ext_vector_type(8)))  __bf16 bf16x8;
typedef __attribute__((ext_vector_type(8)))  float  v8f;

union FragBF { v16bf v; bf16x8 h[2]; };

// ---------------------------------------------------------------------------
// Problem constants
// ---------------------------------------------------------------------------
constexpr int BATCH  = 16384;
constexpr int IN_DIM = 768;
constexpr int E_DIM  = 256;
constexpr int KCB    = 256;   // codes per level
constexpr int NLVL   = 4;

constexpr size_t OUT_N    = (size_t)BATCH * IN_DIM;       // decoder output
constexpr size_t LOSS_OFF = OUT_N;                        // scalar rq_loss
constexpr size_t IDX_OFF  = OUT_N + 1;                    // [B,4] indices (as float)
constexpr size_t SC_OFF   = IDX_OFF + (size_t)BATCH * 4;  // [B,4] min dists
constexpr float  LOSS_SCALE = 1.25f / ((float)BATCH * (float)E_DIM); // (1+beta)/(B*E)

// ---------------------------------------------------------------------------
// Prep kernels
// ---------------------------------------------------------------------------
__global__ void crqvae_f32_to_bf16(const float* __restrict__ in, bf16* __restrict__ out, int n) {
  int i = blockIdx.x * blockDim.x + threadIdx.x;
  if (i < n) out[i] = (bf16)in[i];
}

// Wt[n][k] = (bf16) W[k][n]   (W is [K][N] row-major as in the reference)
__global__ void crqvae_wtrans(const float* __restrict__ w, bf16* __restrict__ wt, int Kd, int Nd) {
  int i = blockIdx.x * blockDim.x + threadIdx.x;
  if (i < Kd * Nd) {
    int k = i / Nd, n = i - k * Nd;
    wt[(size_t)n * Kd + k] = (bf16)w[i];
  }
}

// cbT[l][e][k] = cb[l][k][e];  cbsq[l][k] = sum_e cb[l][k][e]^2
__global__ void crqvae_cb_prep(const float* __restrict__ cb, float* __restrict__ cbT,
                               float* __restrict__ cbsq) {
  int l = blockIdx.x;          // 0..3
  int k = threadIdx.x;         // 0..255
  const float* src = cb + ((size_t)l * KCB + k) * E_DIM;
  float s = 0.f;
  for (int e = 0; e < E_DIM; ++e) {
    float v = src[e];
    s += v * v;
    cbT[((size_t)l * E_DIM + e) * KCB + k] = v;
  }
  cbsq[l * KCB + k] = s;
}

__global__ void crqvae_zero_loss(float* __restrict__ out) {
  if (threadIdx.x == 0 && blockIdx.x == 0) out[LOSS_OFF] = 0.f;
}

// ---------------------------------------------------------------------------
// Tiled bf16 WMMA GEMM:  C[M][N] = act(A[M][K] @ W[K][N] + bias)
// A bf16 row-major, Wt = W^T bf16 [N][K]. Block = 256 threads (8 waves),
// block tile 128(M) x 128(N), K-step 32, double-buffered LDS stages filled with
// gfx1250 GLOBAL_LOAD_ASYNC_TO_LDS_B128 (ASYNCcnt-tracked DMA into LDS).
// Each wave computes a 32x64 tile = 2x4 WMMA fragments = 8 WMMA per K-step.
// The last K-step is peeled so the steady-state loop body is branch-free.
// ---------------------------------------------------------------------------
template <bool RELU, bool OUT_F32>
__global__ __launch_bounds__(256) void crqvae_gemm_wmma(
    const bf16* __restrict__ A, const bf16* __restrict__ Wt,
    const float* __restrict__ bias,
    bf16* __restrict__ Cb, float* __restrict__ Cf,
    int M, int N, int K) {
  __shared__ __align__(16) bf16 As[2][128][40];  // 128x32 + pad (row = 80B, 16B aligned)
  __shared__ __align__(16) bf16 Bs[2][128][40];  // Wt tile: [n][k]

  const int mBase = blockIdx.y * 128;
  const int nBase = blockIdx.x * 128;
  const int tid   = threadIdx.x;
  const int wave  = tid >> 5;
  const int lane  = tid & 31;
  const int wm    = wave >> 1;   // 0..3 -> M sub-tile (32 rows)
  const int wn    = wave & 1;    // 0..1 -> N sub-tile (64 cols)
  const int hh    = lane >> 4;   // half-wave
  const int l16   = lane & 15;

  const v8f zero8 = {0.f, 0.f, 0.f, 0.f, 0.f, 0.f, 0.f, 0.f};
  v8f acc[2][4];
#pragma unroll
  for (int i = 0; i < 2; ++i)
#pragma unroll
    for (int j = 0; j < 4; ++j) acc[i][j] = zero8;

  // Fragment addressing per CDNA5 ISA VGPR layouts (wave32):
  // A 16x32: lanes 0-15 hold K={0..7,16..23}; lanes 16-31 hold K={8..15,24..31}.
  const int ra0 = wm * 32 + l16;
  const int ra1 = ra0 + 16;
  const int ka  = hh ? 8 : 0;
  // B 32x16: lane = N col; lanes 0-15 hold K={0..7,16..23}; lanes 16-31 hold K={8..15,24..31}.
  const int kb  = hh ? 8 : 0;

  // Stage one 128x32 A tile + 128x32 B tile into LDS buffer `buf` using async
  // DMA: 512 16B chunks each, 2 per thread per matrix -> 4 async ops/thread.
  auto stage = [&](int k0, int buf) {
#pragma unroll
    for (int c = 0; c < 2; ++c) {
      int chunk = tid + 256 * c;
      int r  = chunk >> 2;          // 0..127
      int cc = (chunk & 3) * 8;     // 0,8,16,24
      unsigned ldsA = (unsigned)(size_t)&As[buf][r][cc];
      unsigned long long gA = (unsigned long long)(size_t)&A[(size_t)(mBase + r) * K + (k0 + cc)];
      asm volatile("global_load_async_to_lds_b128 %0, %1, off" :: "v"(ldsA), "v"(gA) : "memory");
      unsigned ldsB = (unsigned)(size_t)&Bs[buf][r][cc];
      unsigned long long gB = (unsigned long long)(size_t)&Wt[(size_t)(nBase + r) * K + (k0 + cc)];
      asm volatile("global_load_async_to_lds_b128 %0, %1, off" :: "v"(ldsB), "v"(gB) : "memory");
    }
  };

  // One K-step of fragment loads + 8 WMMA from LDS buffer `buf`.
  auto compute = [&](int buf) {
    FragBF a0, a1, b[4];
    a0.h[0] = *(const bf16x8*)&As[buf][ra0][ka];
    a0.h[1] = *(const bf16x8*)&As[buf][ra0][ka + 16];
    a1.h[0] = *(const bf16x8*)&As[buf][ra1][ka];
    a1.h[1] = *(const bf16x8*)&As[buf][ra1][ka + 16];
#pragma unroll
    for (int j = 0; j < 4; ++j) {
      const int nb = wn * 64 + j * 16 + l16;
      b[j].h[0] = *(const bf16x8*)&Bs[buf][nb][kb];
      b[j].h[1] = *(const bf16x8*)&Bs[buf][nb][kb + 16];
    }
#pragma unroll
    for (int j = 0; j < 4; ++j) {
      acc[0][j] = __builtin_amdgcn_wmma_f32_16x16x32_bf16(false, a0.v, false, b[j].v, (short)0, acc[0][j], false, false);
      acc[1][j] = __builtin_amdgcn_wmma_f32_16x16x32_bf16(false, a1.v, false, b[j].v, (short)0, acc[1][j], false, false);
    }
  };

  const int S = K / 32;
  stage(0, 0);

  // Steady state: branch-free body (wait<=4 ensures this wave's stage-s DMA
  // landed; barrier makes the whole tile visible; trailing barrier protects
  // the buffer that stage s+2 will overwrite).
#pragma unroll 2
  for (int s = 0; s < S - 1; ++s) {
    const int buf = s & 1;
    stage((s + 1) * 32, buf ^ 1);
    asm volatile("s_wait_asynccnt 0x4" ::: "memory");
    __syncthreads();
    compute(buf);
    __syncthreads();
  }
  // Peeled final step.
  asm volatile("s_wait_asynccnt 0x0" ::: "memory");
  __syncthreads();
  compute((S - 1) & 1);

  // Epilogue. C/D layout: vgpr r -> row = tileRow + hh*8 + r, col = tileCol + l16.
  const int row0 = mBase + wm * 32 + hh * 8;   // a0 tiles
  const int row1 = row0 + 16;                  // a1 tiles
#pragma unroll
  for (int j = 0; j < 4; ++j) {
    const int col = nBase + wn * 64 + j * 16 + l16;
    const float bv = bias[col];
#pragma unroll
    for (int r = 0; r < 8; ++r) {
      float v0 = acc[0][j][r] + bv;
      float v1 = acc[1][j][r] + bv;
      if constexpr (RELU) { v0 = fmaxf(v0, 0.f); v1 = fmaxf(v1, 0.f); }
      int r0 = row0 + r, r1 = row1 + r;
      if constexpr (OUT_F32) {
        Cf[(size_t)r0 * N + col] = v0;
        Cf[(size_t)r1 * N + col] = v1;
      } else {
        Cb[(size_t)r0 * N + col] = (bf16)v0;
        Cb[(size_t)r1 * N + col] = (bf16)v1;
      }
    }
  }
}

// ---------------------------------------------------------------------------
// Fused residual VQ: one block (256 thr) per row. Thread k owns code k for the
// distance pass and element k for the residual update. Codebook-loss and
// commitment-loss terms are numerically identical => loss = 1.25*sum(dmin)/(B*E).
// ---------------------------------------------------------------------------
__global__ __launch_bounds__(256) void crqvae_rq(
    const float* __restrict__ z, const float* __restrict__ cb,
    const float* __restrict__ cbT, const float* __restrict__ cbsq,
    bf16* __restrict__ quant, float* __restrict__ out) {
  __shared__ float rsh[256];
  __shared__ float sd[256];
  __shared__ int   si[256];
  const int row = blockIdx.x;
  const int tid = threadIdx.x;

  rsh[tid] = z[(size_t)row * E_DIM + tid];
  float qacc = 0.f;
  float dmins[NLVL];
  int   idxs[NLVL];

  for (int l = 0; l < NLVL; ++l) {
    __syncthreads();
    // ||r||^2
    sd[tid] = rsh[tid] * rsh[tid];
    __syncthreads();
    for (int s = 128; s > 0; s >>= 1) {
      if (tid < s) sd[tid] += sd[tid + s];
      __syncthreads();
    }
    float rnorm = sd[0];
    __syncthreads();

    // d_k = ||r||^2 - 2 r.c_k + ||c_k||^2  (coalesced reads of cbT[l][e][k])
    const float* cbTl = cbT + (size_t)l * E_DIM * KCB;
    float dot = 0.f;
#pragma unroll 8
    for (int e = 0; e < E_DIM; ++e) dot = fmaf(rsh[e], cbTl[(size_t)e * KCB + tid], dot);
    float d = rnorm - 2.f * dot + cbsq[l * KCB + tid];

    // argmin (ties -> lowest index, matching jnp.argmin)
    sd[tid] = d; si[tid] = tid;
    __syncthreads();
    for (int s = 128; s > 0; s >>= 1) {
      if (tid < s && sd[tid + s] < sd[tid]) { sd[tid] = sd[tid + s]; si[tid] = si[tid + s]; }
      __syncthreads();
    }
    int idx = si[0];
    float dmin = sd[0];
    idxs[l] = idx; dmins[l] = dmin;

    float q = cb[((size_t)l * KCB + idx) * E_DIM + tid];
    qacc += q;
    rsh[tid] -= q;
  }

  quant[(size_t)row * E_DIM + tid] = (bf16)qacc;  // straight-through forward value
  if (tid < NLVL) {
    out[IDX_OFF + (size_t)row * NLVL + tid] = (float)idxs[tid];
    out[SC_OFF  + (size_t)row * NLVL + tid] = dmins[tid];
  }
  if (tid == 0) {
    atomicAdd(&out[LOSS_OFF], (dmins[0] + dmins[1] + dmins[2] + dmins[3]) * LOSS_SCALE);
  }
}

// ---------------------------------------------------------------------------
// Host launch
// ---------------------------------------------------------------------------
extern "C" void kernel_launch(void* const* d_in, const int* in_sizes, int n_in,
                              void* d_out, int out_size, void* d_ws, size_t ws_size,
                              hipStream_t stream) {
  (void)in_sizes; (void)n_in; (void)out_size; (void)ws_size;
  const float* x  = (const float*)d_in[0];
  const float* ew[4] = {(const float*)d_in[1], (const float*)d_in[3], (const float*)d_in[5], (const float*)d_in[7]};
  const float* eb[4] = {(const float*)d_in[2], (const float*)d_in[4], (const float*)d_in[6], (const float*)d_in[8]};
  const float* cb = (const float*)d_in[9];
  const float* dw[4] = {(const float*)d_in[10], (const float*)d_in[12], (const float*)d_in[14], (const float*)d_in[16]};
  const float* db[4] = {(const float*)d_in[11], (const float*)d_in[13], (const float*)d_in[15], (const float*)d_in[17]};
  float* out = (float*)d_out;
  char*  ws  = (char*)d_ws;

  const int encD[5] = {768, 2048, 1024, 512, 256};
  const int decD[5] = {256, 512, 1024, 2048, 768};

  size_t off = 0;
  auto alloc = [&](size_t bytes) -> size_t {
    size_t r = off;
    off += (bytes + 255) & ~(size_t)255;
    return r;
  };
  size_t oActA = alloc((size_t)BATCH * 2048 * sizeof(bf16));
  size_t oActB = alloc((size_t)BATCH * 2048 * sizeof(bf16));
  size_t oZ    = alloc((size_t)BATCH * E_DIM * sizeof(float));
  size_t oQ    = alloc((size_t)BATCH * E_DIM * sizeof(bf16));
  size_t oW[8];
  for (int i = 0; i < 4; ++i) oW[i]     = alloc((size_t)encD[i] * encD[i + 1] * sizeof(bf16));
  for (int i = 0; i < 4; ++i) oW[4 + i] = alloc((size_t)decD[i] * decD[i + 1] * sizeof(bf16));
  size_t oCbT = alloc((size_t)NLVL * E_DIM * KCB * sizeof(float));
  size_t oCbS = alloc((size_t)NLVL * KCB * sizeof(float));

  bf16*  actA = (bf16*)(ws + oActA);
  bf16*  actB = (bf16*)(ws + oActB);
  float* z    = (float*)(ws + oZ);
  bf16*  qnt  = (bf16*)(ws + oQ);
  float* cbT  = (float*)(ws + oCbT);
  float* cbS  = (float*)(ws + oCbS);

  // --- prep: weight convert+transpose, x convert, codebook prep, zero loss ---
  for (int i = 0; i < 4; ++i) {
    int n = encD[i] * encD[i + 1];
    crqvae_wtrans<<<(n + 255) / 256, 256, 0, stream>>>(ew[i], (bf16*)(ws + oW[i]), encD[i], encD[i + 1]);
  }
  for (int i = 0; i < 4; ++i) {
    int n = decD[i] * decD[i + 1];
    crqvae_wtrans<<<(n + 255) / 256, 256, 0, stream>>>(dw[i], (bf16*)(ws + oW[4 + i]), decD[i], decD[i + 1]);
  }
  {
    int n = BATCH * IN_DIM;
    crqvae_f32_to_bf16<<<(n + 255) / 256, 256, 0, stream>>>(x, actA, n);
  }
  crqvae_cb_prep<<<NLVL, KCB, 0, stream>>>(cb, cbT, cbS);
  crqvae_zero_loss<<<1, 32, 0, stream>>>(out);

  const dim3 blk(256);
  const int mT = BATCH / 128;

  // --- encoder ---
  crqvae_gemm_wmma<true,  false><<<dim3(2048 / 128, mT), blk, 0, stream>>>(actA, (bf16*)(ws + oW[0]), eb[0], actB, nullptr, BATCH, 2048, 768);
  crqvae_gemm_wmma<true,  false><<<dim3(1024 / 128, mT), blk, 0, stream>>>(actB, (bf16*)(ws + oW[1]), eb[1], actA, nullptr, BATCH, 1024, 2048);
  crqvae_gemm_wmma<true,  false><<<dim3(512  / 128, mT), blk, 0, stream>>>(actA, (bf16*)(ws + oW[2]), eb[2], actB, nullptr, BATCH, 512, 1024);
  crqvae_gemm_wmma<false, true ><<<dim3(256  / 128, mT), blk, 0, stream>>>(actB, (bf16*)(ws + oW[3]), eb[3], nullptr, z, BATCH, 256, 512);

  // --- residual VQ (fused, one block per row) ---
  crqvae_rq<<<BATCH, 256, 0, stream>>>(z, cb, cbT, cbS, qnt, out);

  // --- decoder ---
  crqvae_gemm_wmma<true,  false><<<dim3(512  / 128, mT), blk, 0, stream>>>(qnt,  (bf16*)(ws + oW[4]), db[0], actB, nullptr, BATCH, 512, 256);
  crqvae_gemm_wmma<true,  false><<<dim3(1024 / 128, mT), blk, 0, stream>>>(actB, (bf16*)(ws + oW[5]), db[1], actA, nullptr, BATCH, 1024, 512);
  crqvae_gemm_wmma<true,  false><<<dim3(2048 / 128, mT), blk, 0, stream>>>(actA, (bf16*)(ws + oW[6]), db[2], actB, nullptr, BATCH, 2048, 1024);
  crqvae_gemm_wmma<false, true ><<<dim3(768  / 128, mT), blk, 0, stream>>>(actB, (bf16*)(ws + oW[7]), db[3], nullptr, out, BATCH, 768, 2048);
}
